// MoELayer_36481452213056
// MI455X (gfx1250) — compile-verified, matrix operand order checked by reading, not run on hardware
//
#include <hip/hip_runtime.h>
#include <cstdint>
#include <cstddef>

// ---------------- problem constants ----------------
#define T_TOK 8192      // B*S tokens
#define H_DIM 1024
#define F_DIM 4096
#define E_NUM 8
#define CAP   4096      // per-expert token capacity (expected ~2048)

// GEMM tiling: 8 waves (2x4), each wave computes 2x2 16x16 WMMA tiles
// -> block tile 64 (M) x 128 (N), K-step 32, triple-buffered LDS with a
// 2-deep async-to-LDS prefetch pipeline.
#define BM 64
#define BN 128
#define KT 32
#define NBUF 3
#define LDSS 56         // LDS row stride (elements): 112B rows = 28 dwords,
                        // 16B aligned, bank-conflict-free for 16-lane phases

typedef __attribute__((ext_vector_type(16))) __bf16          v16bf;
typedef __attribute__((ext_vector_type(8)))  float           v8f;
typedef __attribute__((ext_vector_type(8)))  unsigned short  u16x8;
typedef __attribute__((ext_vector_type(16))) unsigned short  u16x16;

// fp32 -> bf16 round-to-nearest-even
__device__ __forceinline__ unsigned short f2bf(float f) {
    unsigned int u = __float_as_uint(f);
    unsigned int r = u + 0x7FFFu + ((u >> 16) & 1u);
    return (unsigned short)(r >> 16);
}

// Build one 16x16x32 bf16 WMMA operand fragment from an LDS row.
// ISA 16-bit A/B layout: lanes 0-15 hold K {0..7,16..23}, lanes 16-31 hold
// K {8..15,24..31} for their row/col -> two 16B chunks per lane.
__device__ __forceinline__ v16bf load_frag(const unsigned short* rowbase, int hi) {
    u16x8 a = *(const u16x8*)(rowbase + hi * 8);
    u16x8 b = *(const u16x8*)(rowbase + 16 + hi * 8);
    u16x16 c = __builtin_shufflevector(a, b, 0,1,2,3,4,5,6,7,8,9,10,11,12,13,14,15);
    return __builtin_bit_cast(v16bf, c);
}

// CDNA5 async global->LDS copy, 16B per lane (GVS mode: SGPR base + VGPR off).
__device__ __forceinline__ void async_b128(unsigned lds_off, unsigned gofs,
                                           unsigned long long base) {
    asm volatile("global_load_async_to_lds_b128 %0, %1, %2"
                 :: "v"(lds_off), "v"(gofs), "s"(base) : "memory");
}
// Wait until at most N async ops remain (in-order completion per ASYNCcnt).
template <int N>
__device__ __forceinline__ void wait_async() {
    asm volatile("s_wait_asynccnt %0" :: "i"(N) : "memory");
}

// ---------------- init: zero out accumulator + expert counts ----------------
__global__ __launch_bounds__(256) void moe_init(float* __restrict__ out, int* __restrict__ counts) {
    size_t i = ((size_t)blockIdx.x * 256 + threadIdx.x) * 4;
    if (i < (size_t)T_TOK * H_DIM) {
        float4 z = make_float4(0.f, 0.f, 0.f, 0.f);
        *(float4*)(out + i) = z;
    }
    if (blockIdx.x == 0 && threadIdx.x < E_NUM) counts[threadIdx.x] = 0;
}

// ---------------- fp32 -> bf16 conversion ----------------
__global__ __launch_bounds__(256) void moe_cvt(const float* __restrict__ in,
                                               unsigned short* __restrict__ out, long n) {
    long i = ((long)blockIdx.x * 256 + threadIdx.x) * 4;
    if (i < n) {
        float4 v = *(const float4*)(in + i);
        ushort4 o;
        o.x = f2bf(v.x); o.y = f2bf(v.y); o.z = f2bf(v.z); o.w = f2bf(v.w);
        *(ushort4*)(out + i) = o;
    }
}

// ---------------- router: logits, top-2 softmax, expert lists ----------------
__global__ __launch_bounds__(256) void moe_router(const float* __restrict__ x,
                                                  const float* __restrict__ gw,
                                                  float* __restrict__ logits,
                                                  int* __restrict__ counts,
                                                  int* __restrict__ tok,
                                                  float* __restrict__ cwbuf) {
    const int t    = blockIdx.x * 8 + (threadIdx.x >> 5);   // one wave32 per token
    const int lane = threadIdx.x & 31;
    const float* xr = x + (size_t)t * H_DIM;

    float acc[E_NUM];
#pragma unroll
    for (int e = 0; e < E_NUM; ++e) acc[e] = 0.f;

    for (int i = lane; i < H_DIM; i += 32) {
        float xv = xr[i];
#pragma unroll
        for (int e = 0; e < E_NUM; ++e) acc[e] += xv * gw[e * H_DIM + i];
    }
#pragma unroll
    for (int e = 0; e < E_NUM; ++e)
        for (int off = 16; off; off >>= 1)
            acc[e] += __shfl_xor(acc[e], off, 32);

    if (lane == 0) {
        int   i0 = 0;   float l0 = acc[0];
#pragma unroll
        for (int e = 1; e < E_NUM; ++e) if (acc[e] > l0) { l0 = acc[e]; i0 = e; }
        int   i1 = -1;  float l1 = -3.4e38f;
#pragma unroll
        for (int e = 0; e < E_NUM; ++e) if (e != i0 && acc[e] > l1) { l1 = acc[e]; i1 = e; }

        float q  = __expf(l1 - l0);          // softmax over the two selected logits
        float w0 = 1.f / (1.f + q);
        float w1 = q * w0;
#pragma unroll
        for (int e = 0; e < E_NUM; ++e) logits[(size_t)t * E_NUM + e] = acc[e];

        int s0 = atomicAdd(&counts[i0], 1);
        if (s0 < CAP) { tok[i0 * CAP + s0] = t; cwbuf[i0 * CAP + s0] = w0; }
        int s1 = atomicAdd(&counts[i1], 1);
        if (s1 < CAP) { tok[i1 * CAP + s1] = t; cwbuf[i1 * CAP + s1] = w1; }
    }
}

// ---------------- routed expert GEMMs (bf16 WMMA, 2-deep async pipeline) -----
// FIRST:  h = gelu(x_gathered @ w1^T)   A = xb[T,H] (gathered), W = w1b[E,F,H]
// !FIRST: out += cw * (h @ w2^T)        A = hbuf[E,CAP,F],      W = w2b[E,H,F]
template <bool FIRST>
__global__ __launch_bounds__(256) void moe_gemm(const unsigned short* __restrict__ Ab,
                                                const unsigned short* __restrict__ Wb,
                                                const int* __restrict__ counts,
                                                const int* __restrict__ tok,
                                                const float* __restrict__ cwbuf,
                                                unsigned short* __restrict__ hout,
                                                float* __restrict__ out) {
    constexpr int KDIM = FIRST ? H_DIM : F_DIM;
    const int e = blockIdx.z;
    int cnt = counts[e]; if (cnt > CAP) cnt = CAP;
    const int mBase = blockIdx.y * BM;
    if (mBase >= cnt) return;
    const int nBase = blockIdx.x * BN;

    __shared__ __align__(16) unsigned short ldsA[NBUF][BM][LDSS];   // ~21 KB
    __shared__ __align__(16) unsigned short ldsB[NBUF][BN][LDSS];   // ~42 KB

    const int tid  = threadIdx.x;
    const int lane = tid & 31;
    const int wave = tid >> 5;
    const int msub = wave >> 2;     // 0..1
    const int nsub = wave & 3;      // 0..3

    // Loader assignments: 16B per lane per request.
    const int arow = tid >> 2;            // 64 rows x 4 chunks
    const int acol = (tid & 3) * 8;
    const int brow = tid >> 2;            // rows brow and brow+64
    const int bcol = (tid & 3) * 8;

    // A source: clamp padded rows to a valid row (masked in epilogue).
    int aRow = mBase + arow; if (aRow >= cnt) aRow = cnt - 1;
    unsigned aoff;
    unsigned long long abase;
    if constexpr (FIRST) {
        int atok = tok[e * CAP + aRow];
        aoff  = (unsigned)atok * H_DIM + (unsigned)acol;
        abase = (unsigned long long)Ab;
    } else {
        aoff  = (unsigned)aRow * F_DIM + (unsigned)acol;
        abase = (unsigned long long)(Ab + (size_t)e * CAP * F_DIM);
    }
    const unsigned long long wbase =
        (unsigned long long)(Wb + (size_t)e * (size_t)F_DIM * H_DIM);
    const unsigned boff0 = (unsigned)(nBase + brow) * KDIM + (unsigned)bcol;
    const unsigned boff1 = boff0 + 64u * KDIM;

    unsigned ldsAo[NBUF], ldsBo0[NBUF], ldsBo1[NBUF];
#pragma unroll
    for (int b = 0; b < NBUF; ++b) {
        ldsAo[b]  = (unsigned)(uintptr_t)&ldsA[b][arow][acol];
        ldsBo0[b] = (unsigned)(uintptr_t)&ldsB[b][brow][bcol];
        ldsBo1[b] = (unsigned)(uintptr_t)&ldsB[b][brow + 64][bcol];
    }

    auto issue = [&](int k, int buf) {        // one tile-group = 3 async ops
        const unsigned k0 = (unsigned)k * KT;
        async_b128(ldsAo[buf],  (aoff  + k0) * 2u, abase);
        async_b128(ldsBo0[buf], (boff0 + k0) * 2u, wbase);
        async_b128(ldsBo1[buf], (boff1 + k0) * 2u, wbase);
    };

    v8f acc00 = {}, acc01 = {}, acc10 = {}, acc11 = {};

    constexpr int NK = KDIM / KT;
    issue(0, 0);
    issue(1, 1);

    const int hi  = lane >> 4;
    const int l15 = lane & 15;
    int cur = 0;
    for (int k = 0; k < NK; ++k) {
        // Oldest in-flight group (buffer `cur`) must be complete; the next
        // group (3 ops) may stay in flight.
        if (k + 1 < NK) wait_async<3>(); else wait_async<0>();
        __syncthreads();   // writes of group k visible; reads of buf (k+2)%3 done
        if (k + 2 < NK) {
            int nxt = cur + 2; if (nxt >= NBUF) nxt -= NBUF;
            issue(k + 2, nxt);
        }

        v16bf a0 = load_frag(&ldsA[cur][msub * 32 +      l15][0], hi);
        v16bf a1 = load_frag(&ldsA[cur][msub * 32 + 16 + l15][0], hi);
        v16bf b0 = load_frag(&ldsB[cur][nsub * 32 +      l15][0], hi);
        v16bf b1 = load_frag(&ldsB[cur][nsub * 32 + 16 + l15][0], hi);

        acc00 = __builtin_amdgcn_wmma_f32_16x16x32_bf16(false, a0, false, b0, (short)0, acc00, false, false);
        acc01 = __builtin_amdgcn_wmma_f32_16x16x32_bf16(false, a0, false, b1, (short)0, acc01, false, false);
        acc10 = __builtin_amdgcn_wmma_f32_16x16x32_bf16(false, a1, false, b0, (short)0, acc10, false, false);
        acc11 = __builtin_amdgcn_wmma_f32_16x16x32_bf16(false, a1, false, b1, (short)0, acc11, false, false);

        cur += 1; if (cur >= NBUF) cur = 0;
    }

    v8f accs[2][2] = { { acc00, acc01 }, { acc10, acc11 } };

#pragma unroll
    for (int mi = 0; mi < 2; ++mi) {
#pragma unroll
        for (int r = 0; r < 8; ++r) {
            const int m = mBase + msub * 32 + mi * 16 + r + 8 * hi;  // C layout: VGPR r -> M
            if constexpr (FIRST) {
#pragma unroll
                for (int ni = 0; ni < 2; ++ni) {
                    const int c = nBase + nsub * 32 + ni * 16 + l15;
                    float v = accs[mi][ni][r];
                    v = 0.5f * v * (1.0f + erff(v * 0.70710678118f));   // exact GELU
                    hout[((size_t)e * CAP + m) * F_DIM + c] = f2bf(v);
                }
            } else {
                if (m < cnt) {
                    const int   t = tok[e * CAP + m];
                    const float w = cwbuf[e * CAP + m];
#pragma unroll
                    for (int ni = 0; ni < 2; ++ni) {
                        const int c = nBase + nsub * 32 + ni * 16 + l15;
                        atomicAdd(&out[(size_t)t * H_DIM + c], w * accs[mi][ni][r]);
                    }
                }
            }
        }
    }
}

// ---------------- launcher ----------------
extern "C" void kernel_launch(void* const* d_in, const int* in_sizes, int n_in,
                              void* d_out, int out_size, void* d_ws, size_t ws_size,
                              hipStream_t stream) {
    const float* x  = (const float*)d_in[0];   // [T,H]
    const float* gw = (const float*)d_in[1];   // [E,H]
    const float* w1 = (const float*)d_in[2];   // [E,F,H]
    const float* w2 = (const float*)d_in[3];   // [E,H,F]
    float* out = (float*)d_out;                // [T,H] out ++ [T,E] logits

    // workspace layout (256B aligned)
    char* ws = (char*)d_ws;
    size_t off = 0;
    auto take = [&](size_t bytes) { char* p = ws + off; off = (off + bytes + 255) & ~(size_t)255; return p; };
    int*            counts = (int*)take(E_NUM * sizeof(int));
    int*            tokl   = (int*)take((size_t)E_NUM * CAP * sizeof(int));
    float*          cwbuf  = (float*)take((size_t)E_NUM * CAP * sizeof(float));
    unsigned short* xb     = (unsigned short*)take((size_t)T_TOK * H_DIM * 2);
    unsigned short* w1b    = (unsigned short*)take((size_t)E_NUM * F_DIM * H_DIM * 2);
    unsigned short* w2b    = (unsigned short*)take((size_t)E_NUM * H_DIM * F_DIM * 2);
    unsigned short* hbuf   = (unsigned short*)take((size_t)E_NUM * CAP * F_DIM * 2);
    (void)ws_size; (void)in_sizes; (void)n_in; (void)out_size;

    // 1) zero output accumulator + counts
    moe_init<<<(T_TOK * (size_t)H_DIM / 4 + 255) / 256, 256, 0, stream>>>(out, counts);

    // 2) fp32 -> bf16 conversions (bf16 weights: 134 MB total -> L2 resident)
    long nx = (long)T_TOK * H_DIM, nw = (long)E_NUM * F_DIM * H_DIM;
    moe_cvt<<<(nx / 4 + 255) / 256, 256, 0, stream>>>(x,  xb,  nx);
    moe_cvt<<<(nw / 4 + 255) / 256, 256, 0, stream>>>(w1, w1b, nw);
    moe_cvt<<<(nw / 4 + 255) / 256, 256, 0, stream>>>(w2, w2b, nw);

    // 3) router: logits + top-2 lists
    moe_router<<<T_TOK / 8, 256, 0, stream>>>(x, gw, out + (size_t)T_TOK * H_DIM,
                                              counts, tokl, cwbuf);

    // 4) h = gelu(x @ w1^T) per expert (gathered)
    dim3 g1(F_DIM / BN, CAP / BM, E_NUM);
    moe_gemm<true><<<g1, 256, 0, stream>>>(xb, w1b, counts, tokl, cwbuf, hbuf, out);

    // 5) out += cw * (h @ w2^T)
    dim3 g2(H_DIM / BN, CAP / BM, E_NUM);
    moe_gemm<false><<<g2, 256, 0, stream>>>(hbuf, w2b, counts, tokl, cwbuf, hbuf, out);
}
